// MultiHeadAttention_79748952752781
// MI455X (gfx1250) — compile-verified, matrix operand order checked by reading, not run on hardware
//
#include <hip/hip_runtime.h>
#include <hip/hip_bf16.h>

// Problem constants (match reference)
#define Bv   4
#define Sv   1024
#define Dv   1024
#define Hv   16
#define DKv  64
#define SCALEv 0.125f        // 1/sqrt(DK)
#define NEGv  (-1.0e9f)

typedef __bf16 bf16_t;
typedef __attribute__((ext_vector_type(16))) __bf16 v16bf;
typedef __attribute__((ext_vector_type(8)))  __bf16 v8bf;
typedef __attribute__((ext_vector_type(8)))  float  v8f;

__device__ __forceinline__ v8f wmma_bf16(v16bf a, v16bf b, v8f c) {
  // D = A(16x32) * B(32x16) + C, fp32 accumulate
  return __builtin_amdgcn_wmma_f32_16x16x32_bf16(false, a, false, b, (short)0, c, false, false);
}

// Load a 16x32 bf16 fragment from a row-major [16][ld] tile (global or LDS).
// Per ISA layout: lanes 0-15 hold K 0..7 & 16..23; lanes 16-31 hold K 8..15 & 24..31.
__device__ __forceinline__ v16bf load_frag(const bf16_t* p, int ld) {
  const int lane = threadIdx.x & 31;
  const int r    = lane & 15;
  const int hh   = (lane >> 4) << 3;   // 0 or 8
  const bf16_t* q = p + (size_t)r * ld + hh;
  union { v16bf v; v8bf h[2]; } u;
  u.h[0] = *reinterpret_cast<const v8bf*>(q);        // K = hh .. hh+7
  u.h[1] = *reinterpret_cast<const v8bf*>(q + 16);   // K = 16+hh .. 16+hh+7
  return u.v;
}

__device__ __forceinline__ float half_reduce_max(float v) {
  v = fmaxf(v, __shfl_xor(v, 1, 32));
  v = fmaxf(v, __shfl_xor(v, 2, 32));
  v = fmaxf(v, __shfl_xor(v, 4, 32));
  v = fmaxf(v, __shfl_xor(v, 8, 32));
  return v;
}
__device__ __forceinline__ float half_reduce_sum(float v) {
  v += __shfl_xor(v, 1, 32);
  v += __shfl_xor(v, 2, 32);
  v += __shfl_xor(v, 4, 32);
  v += __shfl_xor(v, 8, 32);
  return v;
}

// 16-byte async global->LDS copy (gfx1250 GLOBAL_LOAD_ASYNC_TO_LDS_B128, ASYNCcnt)
__device__ __forceinline__ void async_copy16(unsigned lds_off, unsigned long long gaddr) {
  asm volatile("global_load_async_to_lds_b128 %0, %1, off"
               :: "v"(lds_off), "v"(gaddr) : "memory");
}
__device__ __forceinline__ unsigned lds_byte_off(const void* p) {
  // generic LDS pointer: low 32 bits are the LDS byte offset
  return (unsigned)(size_t)p;
}

// ---------------- conversion kernels ----------------
__global__ void k_cvt(const float* __restrict__ src, bf16_t* __restrict__ dst, int n) {
  for (int i = blockIdx.x * blockDim.x + threadIdx.x; i < n; i += gridDim.x * blockDim.x)
    dst[i] = (bf16_t)src[i];
}

// w: [M][K][N] fp32  ->  wt: [M][N][K] bf16
__global__ void k_tcvt(const float* __restrict__ w, bf16_t* __restrict__ wt,
                       int M, int K, int N) {
  long total = (long)M * K * N;
  for (long i = blockIdx.x * (long)blockDim.x + threadIdx.x; i < total;
       i += (long)gridDim.x * blockDim.x) {
    int n = (int)(i % N);
    long t = i / N;
    int k = (int)(t % K);
    int m = (int)(t / K);
    wt[((long)m * N + n) * K + k] = (bf16_t)w[i];
  }
}

// ---------------- fused QKV projection ----------------
// One wave computes a 32x64 strip (32 rows of s, full DK of one head) of one of
// {Q, K, V}: 2 A-frags + 4 B-frags -> 8 WMMAs per K-step. V written transposed.
__global__ void __launch_bounds__(256)
k_qkv(const bf16_t* __restrict__ xb,
      const bf16_t* __restrict__ wqT, const bf16_t* __restrict__ wkT,
      const bf16_t* __restrict__ wvT,
      bf16_t* __restrict__ qb, bf16_t* __restrict__ kb, bf16_t* __restrict__ vTb) {
  const int wave  = (blockIdx.x * blockDim.x + threadIdx.x) >> 5;  // 0..6143
  const int lane  = threadIdx.x & 31;
  const int mat   = wave >> 11;          // 0=Q 1=K 2=V (2048 waves each)
  const int r     = wave & 2047;
  const int bh    = r >> 5;              // S/32 = 32 strips per (b,h)
  const int strip = r & 31;
  const int b     = bh >> 4;
  const int head  = bh & 15;

  const bf16_t* w  = (mat == 0) ? wqT : (mat == 1) ? wkT : wvT;
  const bf16_t* A0 = xb + ((size_t)b * Sv + strip * 32) * Dv;
  const bf16_t* B0 = w + (size_t)head * DKv * Dv;

  v8f zero = {};
  v8f acc[2][4];
#pragma unroll
  for (int m = 0; m < 2; ++m)
#pragma unroll
    for (int j = 0; j < 4; ++j) acc[m][j] = zero;

  for (int kk = 0; kk < Dv; kk += 32) {
    v16bf a0 = load_frag(A0 + kk, Dv);
    v16bf a1 = load_frag(A0 + (size_t)16 * Dv + kk, Dv);
#pragma unroll
    for (int j = 0; j < 4; ++j) {
      v16bf bb = load_frag(B0 + (size_t)(16 * j) * Dv + kk, Dv);
      acc[0][j] = wmma_bf16(a0, bb, acc[0][j]);
      acc[1][j] = wmma_bf16(a1, bb, acc[1][j]);
    }
  }

  const int col = lane & 15;
  const int ro  = (lane >> 4) * 8;
  if (mat < 2) {
    bf16_t* o = ((mat == 0) ? qb : kb) +
                ((size_t)(b * Hv + head) * Sv + strip * 32) * DKv;
#pragma unroll
    for (int m = 0; m < 2; ++m)
#pragma unroll
      for (int j = 0; j < 4; ++j)
#pragma unroll
        for (int i = 0; i < 8; ++i)
          o[(size_t)(m * 16 + ro + i) * DKv + j * 16 + col] = (bf16_t)acc[m][j][i];
  } else {
    // vT[b][h][dk][t]
    bf16_t* o = vTb + ((size_t)(b * Hv + head) * DKv) * Sv + strip * 32;
#pragma unroll
    for (int m = 0; m < 2; ++m)
#pragma unroll
      for (int j = 0; j < 4; ++j)
#pragma unroll
        for (int i = 0; i < 8; ++i)
          o[(size_t)(j * 16 + col) * Sv + m * 16 + ro + i] = (bf16_t)acc[m][j][i];
  }
}

// ---------------- flash-style attention ----------------
// Block = 8 waves sharing (b, head); wave w owns query rows [st*16, st*16+16).
// K/V tiles for each 32-wide t-chunk are staged into LDS once per block with
// async global->LDS copies (double buffered), then consumed by all 8 waves.
__global__ void __launch_bounds__(256)
k_attn(const bf16_t* __restrict__ qb, const bf16_t* __restrict__ kb,
       const bf16_t* __restrict__ vTb, const unsigned char* __restrict__ mask,
       bf16_t* __restrict__ ctxb) {
  __shared__ __align__(16) bf16_t lds_k[2][32 * 64];  // [t 32][dk 64]
  __shared__ __align__(16) bf16_t lds_v[2][64 * 32];  // [dk 64][t 32]
  __shared__ __align__(16) bf16_t plds[8][16 * 32];   // per-wave P slab

  const int tid  = threadIdx.x;
  const int wid  = tid >> 5;
  const int lane = tid & 31;
  const int gw   = blockIdx.x * 8 + wid;
  const int st   = gw & 63;
  const int bh   = gw >> 6;          // same for all waves in the block
  const int head = bh & 15;
  const int b    = bh >> 4;

  const bf16_t* qbase = qb + ((size_t)(b * Hv + head) * Sv + st * 16) * DKv;
  const bf16_t* kbh   = kb + ((size_t)(b * Hv + head) * Sv) * DKv;
  const bf16_t* vbh   = vTb + ((size_t)(b * Hv + head) * DKv) * Sv;
  const unsigned char* mrow = mask + ((size_t)b * Sv + st * 16) * Sv;

  // Stage one 32-t chunk: K tile is 4KB contiguous; V^T tile is 64 strided rows.
  // 256 threads x 16B each per tile -> each wave issues exactly 2 async loads.
  auto stage = [&](int t0, int buf) {
    async_copy16(lds_byte_off(&lds_k[buf][tid * 8]),
                 (unsigned long long)(size_t)(kbh + (size_t)t0 * DKv + tid * 8));
    const int vrow = tid >> 2, vseg = tid & 3;
    async_copy16(lds_byte_off(&lds_v[buf][vrow * 32 + vseg * 8]),
                 (unsigned long long)(size_t)(vbh + (size_t)vrow * Sv + t0 + vseg * 8));
  };

  const v16bf aq0 = load_frag(qbase, DKv);
  const v16bf aq1 = load_frag(qbase + 32, DKv);

  float rm[8], rs[8];
  v8f acc0 = {}, acc1 = {}, acc2 = {}, acc3 = {};
#pragma unroll
  for (int i = 0; i < 8; ++i) { rm[i] = -3.0e38f; rs[i] = 0.0f; }

  const int col = lane & 15;
  const int ro  = (lane >> 4) * 8;
  bf16_t* myp = &plds[wid][0];

  stage(0, 0);
  for (int t0 = 0; t0 < Sv; t0 += 32) {
    const int buf = (t0 >> 5) & 1;
    const int tn  = (t0 + 32 < Sv) ? (t0 + 32) : 0;  // wrap keeps wait count uniform
    stage(tn, buf ^ 1);
    // async loads complete in order: <=2 outstanding => current buffer done
    asm volatile("s_wait_asynccnt 2" ::: "memory");
    __syncthreads();

    const bf16_t* kl = &lds_k[buf][0];
    const bf16_t* vl = &lds_v[buf][0];
    v16bf bk0a = load_frag(kl, 64);
    v16bf bk0b = load_frag(kl + 32, 64);
    v16bf bk1a = load_frag(kl + 16 * 64, 64);
    v16bf bk1b = load_frag(kl + 16 * 64 + 32, 64);

    v8f s0 = {}; s0 = wmma_bf16(aq0, bk0a, s0); s0 = wmma_bf16(aq1, bk0b, s0);
    v8f s1 = {}; s1 = wmma_bf16(aq0, bk1a, s1); s1 = wmma_bf16(aq1, bk1b, s1);

#pragma unroll
    for (int i = 0; i < 8; ++i) {
      const int row = ro + i;
      float a = s0[i] * SCALEv;
      float c = s1[i] * SCALEv;
      if (!mrow[(size_t)row * Sv + t0 + col])      a = NEGv;
      if (!mrow[(size_t)row * Sv + t0 + 16 + col]) c = NEGv;
      const float tm = half_reduce_max(fmaxf(a, c));
      const float nm = fmaxf(rm[i], tm);
      const float corr = __expf(rm[i] - nm);
      rm[i] = nm;
      const float pa = __expf(a - nm);
      const float pc = __expf(c - nm);
      rs[i] = rs[i] * corr + half_reduce_sum(pa + pc);
      acc0[i] *= corr; acc1[i] *= corr; acc2[i] *= corr; acc3[i] *= corr;
      myp[row * 32 + col]      = (bf16_t)pa;
      myp[row * 32 + 16 + col] = (bf16_t)pc;
    }
    // wave-private LDS slab: order stores before cross-lane fragment reload
    asm volatile("s_wait_dscnt 0" ::: "memory");
    const v16bf pf = load_frag(myp, 32);
    acc0 = wmma_bf16(pf, load_frag(vl, 32), acc0);
    acc1 = wmma_bf16(pf, load_frag(vl + 16 * 32, 32), acc1);
    acc2 = wmma_bf16(pf, load_frag(vl + 32 * 32, 32), acc2);
    acc3 = wmma_bf16(pf, load_frag(vl + 48 * 32, 32), acc3);

    __syncthreads();  // everyone done reading buf before next stage overwrites it
  }

  // epilogue: ctx / rowsum -> concat layout [b][s][head*64 + dk]
  bf16_t* o = ctxb + ((size_t)b * Sv + st * 16) * Dv + head * DKv;
#pragma unroll
  for (int i = 0; i < 8; ++i) {
    const float inv = 1.0f / rs[i];
    const size_t rbase = (size_t)(ro + i) * Dv;
    o[rbase + 0  + col] = (bf16_t)(acc0[i] * inv);
    o[rbase + 16 + col] = (bf16_t)(acc1[i] * inv);
    o[rbase + 32 + col] = (bf16_t)(acc2[i] * inv);
    o[rbase + 48 + col] = (bf16_t)(acc3[i] * inv);
  }
}

// ---------------- output projection + residual + LN partials ----------------
// One wave computes a 32x64 output strip: 8 WMMAs per K-step.
__global__ void __launch_bounds__(256)
k_oproj(const bf16_t* __restrict__ ctxb, const bf16_t* __restrict__ woT,
        const float* __restrict__ x, float* __restrict__ outp,
        float* __restrict__ part) {
  const int wave = (blockIdx.x * blockDim.x + threadIdx.x) >> 5;  // 0..2047
  const int lane = threadIdx.x & 31;
  const int colstrip = wave & 15;        // D/64 = 16
  const int rowstrip = wave >> 4;        // (B*S)/32 = 128

  const bf16_t* A0 = ctxb + (size_t)rowstrip * 32 * Dv;
  const bf16_t* B0 = woT  + (size_t)colstrip * 64 * Dv;

  v8f zero = {};
  v8f acc[2][4];
#pragma unroll
  for (int m = 0; m < 2; ++m)
#pragma unroll
    for (int j = 0; j < 4; ++j) acc[m][j] = zero;

  for (int kk = 0; kk < Dv; kk += 32) {
    v16bf a0 = load_frag(A0 + kk, Dv);
    v16bf a1 = load_frag(A0 + (size_t)16 * Dv + kk, Dv);
#pragma unroll
    for (int j = 0; j < 4; ++j) {
      v16bf bb = load_frag(B0 + (size_t)(16 * j) * Dv + kk, Dv);
      acc[0][j] = wmma_bf16(a0, bb, acc[0][j]);
      acc[1][j] = wmma_bf16(a1, bb, acc[1][j]);
    }
  }

  const int col = lane & 15;
  const int ro  = (lane >> 4) * 8;
  float sum = 0.0f, sq = 0.0f;
#pragma unroll
  for (int m = 0; m < 2; ++m)
#pragma unroll
    for (int j = 0; j < 4; ++j)
#pragma unroll
      for (int i = 0; i < 8; ++i) {
        const size_t idx = (size_t)(rowstrip * 32 + m * 16 + ro + i) * Dv +
                           colstrip * 64 + j * 16 + col;
        const float v = acc[m][j][i] + x[idx];
        outp[idx] = v;
        sum += v; sq += v * v;
      }
#pragma unroll
  for (int msk = 1; msk < 32; msk <<= 1) {
    sum += __shfl_xor(sum, msk, 32);
    sq  += __shfl_xor(sq,  msk, 32);
  }
  if (lane == 0) { part[(size_t)wave * 2] = sum; part[(size_t)wave * 2 + 1] = sq; }
}

// ---------------- deterministic stats reduction (per batch) ----------------
__global__ void k_stats(const float* __restrict__ part, float* __restrict__ stats) {
  __shared__ float ss[256], sq[256];
  const int b = blockIdx.x, t = threadIdx.x;
  const float* p = part + (size_t)b * 512 * 2;   // 512 strips per batch
  float s = 0.0f, q = 0.0f;
  for (int i = t; i < 512; i += 256) { s += p[(size_t)i * 2]; q += p[(size_t)i * 2 + 1]; }
  ss[t] = s; sq[t] = q;
  __syncthreads();
  for (int off = 128; off > 0; off >>= 1) {
    if (t < off) { ss[t] += ss[t + off]; sq[t] += sq[t + off]; }
    __syncthreads();
  }
  if (t == 0) {
    const float n  = (float)Sv * (float)Dv;
    const float mu = ss[0] / n;
    const float var = sq[0] / n - mu * mu;
    stats[b * 2]     = mu;
    stats[b * 2 + 1] = rsqrtf(var + 1e-5f);
  }
}

__global__ void k_ln(const float* __restrict__ outp, const float* __restrict__ stats,
                     float* __restrict__ out, int n) {
  const int sd = Sv * Dv;
  for (int i = blockIdx.x * blockDim.x + threadIdx.x; i < n; i += gridDim.x * blockDim.x) {
    const int b = i / sd;
    out[i] = (outp[i] - stats[b * 2]) * stats[b * 2 + 1];
  }
}

extern "C" void kernel_launch(void* const* d_in, const int* in_sizes, int n_in,
                              void* d_out, int out_size, void* d_ws, size_t ws_size,
                              hipStream_t stream) {
  (void)in_sizes; (void)n_in; (void)out_size; (void)ws_size;
  const unsigned char* mask = (const unsigned char*)d_in[0];
  const float* x  = (const float*)d_in[1];
  const float* wq = (const float*)d_in[2];
  const float* wk = (const float*)d_in[3];
  const float* wv = (const float*)d_in[4];
  const float* wo = (const float*)d_in[5];
  float* out = (float*)d_out;

  char* ws = (char*)d_ws;
  size_t off = 0;
  auto alloc = [&](size_t bytes) -> char* {
    char* p = ws + off;
    off += (bytes + 255) & ~(size_t)255;
    return p;
  };

  bf16_t* xb   = (bf16_t*)alloc((size_t)Bv * Sv * Dv * 2);          // 8 MB
  bf16_t* wqT  = (bf16_t*)alloc((size_t)Hv * DKv * Dv * 2);         // 2 MB
  bf16_t* wkT  = (bf16_t*)alloc((size_t)Hv * DKv * Dv * 2);
  bf16_t* wvT  = (bf16_t*)alloc((size_t)Hv * DKv * Dv * 2);
  bf16_t* woT  = (bf16_t*)alloc((size_t)Dv * Dv * 2);               // 2 MB
  bf16_t* qb   = (bf16_t*)alloc((size_t)Bv * Hv * Sv * DKv * 2);    // 8 MB
  bf16_t* kbuf = (bf16_t*)alloc((size_t)Bv * Hv * Sv * DKv * 2);
  bf16_t* vTb  = (bf16_t*)alloc((size_t)Bv * Hv * Sv * DKv * 2);
  bf16_t* ctxb = (bf16_t*)alloc((size_t)Bv * Sv * Dv * 2);
  float*  outp = (float*)alloc((size_t)Bv * Sv * Dv * 4);           // 16 MB
  float*  part = (float*)alloc((size_t)2048 * 2 * 4);
  float*  stat = (float*)alloc((size_t)Bv * 2 * 4);

  k_cvt<<<4096, 256, 0, stream>>>(x, xb, Bv * Sv * Dv);
  k_tcvt<<<1024, 256, 0, stream>>>(wq, wqT, Hv, Dv, DKv);
  k_tcvt<<<1024, 256, 0, stream>>>(wk, wkT, Hv, Dv, DKv);
  k_tcvt<<<1024, 256, 0, stream>>>(wv, wvT, Hv, Dv, DKv);
  k_tcvt<<<1024, 256, 0, stream>>>(wo, woT, 1, Dv, Dv);

  // 3 * B*H*(S/32) = 6144 waves, 8 waves/block
  k_qkv<<<768, 256, 0, stream>>>(xb, wqT, wkT, wvT, qb, kbuf, vTb);

  // B*H*(S/16) = 4096 waves, 8 waves/block sharing (b,head)
  k_attn<<<512, 256, 0, stream>>>(qb, kbuf, vTb, mask, ctxb);

  // (B*S/32)*(D/64) = 2048 waves
  k_oproj<<<256, 256, 0, stream>>>(ctxb, woT, x, outp, part);

  k_stats<<<Bv, 256, 0, stream>>>(part, stat);
  k_ln<<<4096, 256, 0, stream>>>(outp, stat, out, Bv * Sv * Dv);
}
